// CIN_35399120453661
// MI455X (gfx1250) — matrix operand order repaired and checked by reading, not compiled, not run on hardware
//
#include <hip/hip_runtime.h>
#include <hip/hip_bf16.h>

typedef __attribute__((ext_vector_type(16))) _Float16 v16h;
typedef __attribute__((ext_vector_type(8)))  _Float16 h8;
typedef __attribute__((ext_vector_type(8)))  float    v8f;

#define BLK 128   // 4 waves of 32

static __device__ __forceinline__ v16h mkv16(h8 a, h8 b) {
    return __builtin_shufflevector(a, b, 0,1,2,3,4,5,6,7,8,9,10,11,12,13,14,15);
}

// GVS-mode async copy: LDS[vdst + imm] = MEM[saddr + vaddr + imm], 16B per lane,
// tracked by ASYNCcnt (CDNA5 ISA 10. / 15.18.3, opcode 98).
#define ASYNC_LD16(ldsv, goff, sbase, imm)                                   \
    asm volatile("global_load_async_to_lds_b128 %0, %1, %2 offset:" #imm     \
                 :: "v"(ldsv), "v"(goff), "s"(sbase) : "memory")

#define WAIT_ASYNC0() asm volatile("s_wait_asynccnt 0x0" ::: "memory")

static __device__ __forceinline__ unsigned lds_addr_u32(const void* p) {
    // low 32 bits of a flat shared pointer == LDS byte address (aperture rule)
    return (unsigned)(unsigned long long)p;
}

// ---------------------------------------------------------------------------
// Pre-pass: convert W (f32, [40][Horig][128]) to f16, pre-swizzled into the
// per-lane WMMA B-fragment layout, split into two 16B planes per lane so the
// main loop's ds_load_b128 reads are bank-conflict-free:
//   frag (ks,lt) occupies 512 halves; plane0 at lane*8 (elems K..K+7),
//   plane1 at 256+lane*8 (elems K+8..K+15), K = ks*32 + (lane>>4)*16.
// Layer 0 pads H 40->64 with zeros so the main loop stays uniform.
// ---------------------------------------------------------------------------
__global__ void cin_swizzle_w(const float* __restrict__ W0,
                              const float* __restrict__ W1,
                              const float* __restrict__ W2,
                              _Float16* __restrict__ o0,
                              _Float16* __restrict__ o1,
                              _Float16* __restrict__ o2)
{
    const int layer = blockIdx.y;
    const float* W; _Float16* o; int hshift, horig, kpad;
    if (layer == 0)      { W = W0; o = o0; hshift = 6; horig = 40;  kpad = 2560; }
    else if (layer == 1) { W = W1; o = o1; hshift = 7; horig = 128; kpad = 5120; }
    else                 { W = W2; o = o2; hshift = 7; horig = 128; kpad = 5120; }

    const int t = blockIdx.x * blockDim.x + threadIdx.x;
    if (t >= kpad * 8) return;                 // one thread per (ks, lt, lane)

    const int lane = t & 31;
    const int lt   = (t >> 5) & 7;
    const int ks   = t >> 8;
    const int l    = lt * 16 + (lane & 15);
    const int kb   = ks * 32 + (lane >> 4) * 16;
    const int hmask = (1 << hshift) - 1;

    h8 lo, hi;
#pragma unroll
    for (int e = 0; e < 8; ++e) {
        int K = kb + e, m = K >> hshift, h = K & hmask;
        lo[e] = (_Float16)((h < horig) ? W[(m * horig + h) * 128 + l] : 0.0f);
    }
#pragma unroll
    for (int e = 0; e < 8; ++e) {
        int K = kb + 8 + e, m = K >> hshift, h = K & hmask;
        hi[e] = (_Float16)((h < horig) ? W[(m * horig + h) * 128 + l] : 0.0f);
    }
    const size_t fb = (size_t)(ks * 8 + lt) * 512;
    *(h8*)(o + fb + lane * 8)       = lo;
    *(h8*)(o + fb + 256 + lane * 8) = hi;
}

// ---------------------------------------------------------------------------
// One CIN layer for one batch element (whole block cooperates).
//   C[dd,l] = sum_k  x0T[dd][k>>hkshift] * xkT[dd][k & mask] * Wsw[k][l]
// Wave dt owns d-rows [dt*16, dt*16+16); 8 l-tiles accumulated per wave.
// The 8KB W slab per K-step is double-buffered in LDS via async-to-LDS DMA.
// ---------------------------------------------------------------------------
static __device__ __forceinline__ void cin_layer(
    const _Float16* __restrict__ wsw,   // swizzled f16 weights (global)
    int nsteps, int hkshift, int xstride,
    const _Float16* xkT,                // LDS: [64][xstride] f16 (transposed xk)
    _Float16* xnT,                      // LDS: [64][144] f16 output (or null)
    const _Float16* x0T,                // LDS: [64][72] f16 (transposed x0, padded)
    _Float16* wbuf0, _Float16* wbuf1,   // LDS: 2 x 4096 halves (8KB slabs)
    float* prow,                        // LDS: 128 floats (this layer's p), zeroed
    int dt, int lane, int tid)
{
    v8f acc[8];
    const v8f vzero = {0.f,0.f,0.f,0.f,0.f,0.f,0.f,0.f};
#pragma unroll
    for (int lt = 0; lt < 8; ++lt) acc[lt] = vzero;

    const int dd    = dt * 16 + (lane & 15);     // A-row this lane provides
    const int hof   = (lane >> 4) * 16;          // K-run offset within step
    const int hmask = (1 << hkshift) - 1;

    const unsigned ldsB0 = lds_addr_u32(wbuf0) + (unsigned)tid * 16;
    const unsigned ldsB1 = lds_addr_u32(wbuf1) + (unsigned)tid * 16;

    // ---- prologue: async-stage K-step 0 into wbuf0 ----
    {
        const unsigned g0 = (unsigned)tid * 16;
        ASYNC_LD16(ldsB0, g0, wsw, 0);
        ASYNC_LD16(ldsB0, g0, wsw, 2048);
        ASYNC_LD16(ldsB0, g0, wsw, 4096);
        ASYNC_LD16(ldsB0, g0, wsw, 6144);
    }
    WAIT_ASYNC0();
    __syncthreads();

    for (int st = 0; st < nsteps; ++st) {
        const _Float16* cur = (st & 1) ? wbuf1 : wbuf0;

        // kick off async DMA of the next slab into the other buffer
        if (st + 1 < nsteps) {
            const unsigned ldsn = ((st + 1) & 1) ? ldsB1 : ldsB0;
            const unsigned gn   = (unsigned)(st + 1) * 8192u + (unsigned)tid * 16;
            ASYNC_LD16(ldsn, gn, wsw, 0);
            ASYNC_LD16(ldsn, gn, wsw, 2048);
            ASYNC_LD16(ldsn, gn, wsw, 4096);
            ASYNC_LD16(ldsn, gn, wsw, 6144);
        }

        const int kb = st * 32;
        const int m  = kb >> hkshift;            // constant across this step
        const int hb = kb & hmask;
        const _Float16 s = x0T[dd * 72 + m];     // x0 scalar for this lane

        // Build A fragment: two contiguous 8-element K-runs, scaled by s
        const _Float16* xr = xkT + dd * xstride + hb + hof;
        const v16h av = mkv16(*(const h8*)xr, *(const h8*)(xr + 8)) * s;

        // 8 WMMAs: one per l-tile, B fragments from the LDS slab (2 planes)
#pragma unroll
        for (int lt = 0; lt < 8; ++lt) {
            const _Float16* wp = cur + (lt << 9) + (lane << 3);
            const v16h bv = mkv16(*(const h8*)wp, *(const h8*)(wp + 256));
            acc[lt] = __builtin_amdgcn_wmma_f32_16x16x32_f16(
                false, av, false, bv, (short)0, acc[lt], false, false);
        }

        WAIT_ASYNC0();        // our DMA into the other buffer has landed
        __syncthreads();      // everyone done reading cur / DMA visible
    }

    // ---- epilogue: p row-sums + f16 x_next into LDS for the next layer ----
    const int ncol  = lane & 15;
    const int rbase = dt * 16 + (lane >> 4) * 8;
#pragma unroll
    for (int lt = 0; lt < 8; ++lt) {
        const int lcol = lt * 16 + ncol;
        float ps = 0.f;
#pragma unroll
        for (int r = 0; r < 8; ++r) {
            const float v = acc[lt][r];          // C[rbase+r][lcol]
            ps += v;
            if (xnT) xnT[(rbase + r) * 144 + lcol] = (_Float16)v;
        }
        ps += __shfl_xor(ps, 16, 32);            // combine d-halves (same column)
        if (lane < 16) atomicAdd(&prow[lcol], ps);
    }
    __syncthreads();
}

// ---------------------------------------------------------------------------
// Main kernel: one batch element per block, 128 threads (4 wave32).
// ---------------------------------------------------------------------------
__global__ __launch_bounds__(BLK) void cin_main(
    const float* __restrict__ x,
    const _Float16* __restrict__ w0sw,
    const _Float16* __restrict__ w1sw,
    const _Float16* __restrict__ w2sw,
    float* __restrict__ out)
{
    __shared__ _Float16 x0T[64 * 72];        // x0^T [dd][m], stride 72 (pad: banks)
    __shared__ _Float16 xkA[64 * 144];       // x1^T [dd][l], stride 144
    __shared__ _Float16 xkB[64 * 144];       // x2^T [dd][l], stride 144
    __shared__ _Float16 wbuf0[4096];         // W slab double buffer (8KB each)
    __shared__ _Float16 wbuf1[4096];
    __shared__ float    pbuf[384];

    const int tid  = threadIdx.x;
    const int b    = blockIdx.x;
    const int lane = tid & 31;
    const int dt   = tid >> 5;

    // Load x[b] (40x64 f32), convert to f16, transpose into x0T, zero-pad m>=40
    const float* xb = x + (size_t)b * 40 * 64;
    for (int i = tid; i < 64 * 64; i += BLK) {
        const int m = i >> 6, dd = i & 63;
        const float v = (m < 40) ? xb[m * 64 + dd] : 0.0f;
        x0T[dd * 72 + m] = (_Float16)v;
    }
    for (int i = tid; i < 384; i += BLK) pbuf[i] = 0.0f;
    __syncthreads();

    // Layer 0: xk = x0 (H padded to 64, K = 2560, 80 steps)
    cin_layer(w0sw,  80, 6, 72,  x0T, xkA, x0T, wbuf0, wbuf1, pbuf +   0, dt, lane, tid);
    // Layer 1: xk = x1 (H = 128, K = 5120, 160 steps)
    cin_layer(w1sw, 160, 7, 144, xkA, xkB, x0T, wbuf0, wbuf1, pbuf + 128, dt, lane, tid);
    // Layer 2: xk = x2 (no x_next needed)
    cin_layer(w2sw, 160, 7, 144, xkB, (_Float16*)nullptr, x0T, wbuf0, wbuf1, pbuf + 256, dt, lane, tid);

    for (int i = tid; i < 384; i += BLK) out[(size_t)b * 384 + i] = pbuf[i];
}

// ---------------------------------------------------------------------------
extern "C" void kernel_launch(void* const* d_in, const int* in_sizes, int n_in,
                              void* d_out, int out_size, void* d_ws, size_t ws_size,
                              hipStream_t stream)
{
    const float* x  = (const float*)d_in[0];   // [2048, 40, 64]
    const float* W0 = (const float*)d_in[1];   // [40, 40, 128]
    const float* W1 = (const float*)d_in[2];   // [40, 128, 128]
    const float* W2 = (const float*)d_in[3];   // [40, 128, 128]

    _Float16* ws = (_Float16*)d_ws;
    _Float16* o0 = ws;                         // 2560*128 halves (padded W0)
    _Float16* o1 = o0 + 2560 * 128;            // 5120*128 halves
    _Float16* o2 = o1 + 5120 * 128;            // 5120*128 halves

    dim3 gsw(160, 3);                          // covers max 40960 threads/layer
    cin_swizzle_w<<<gsw, 256, 0, stream>>>(W0, W1, W2, o0, o1, o2);

    cin_main<<<2048, BLK, 0, stream>>>(x, o0, o1, o2, (float*)d_out);
}
